// TransformerEncoderBlock_86139864089200
// MI455X (gfx1250) — compile-verified, matrix operand order checked by reading, not run on hardware
//
#include <hip/hip_runtime.h>
#include <hip/hip_bf16.h>

// Problem constants (match reference)
#define Dm   1024
#define Hh   16
#define HDd  64
#define FFf  4096
#define Bb   2
#define Ss   2048
#define EPSc 1e-5f

typedef __attribute__((ext_vector_type(16))) __bf16 v16bf;
typedef __attribute__((ext_vector_type(8)))  __bf16 v8bf;
typedef __attribute__((ext_vector_type(8)))  float  v8f;
typedef unsigned int u32x4 __attribute__((ext_vector_type(4)));
typedef int          i32x4 __attribute__((ext_vector_type(4)));
typedef int          i32x8 __attribute__((ext_vector_type(8)));

#ifndef __has_builtin
#define __has_builtin(x) 0
#endif
#if __has_builtin(__builtin_amdgcn_tensor_load_to_lds) && \
    __has_builtin(__builtin_amdgcn_s_wait_tensorcnt)
#define USE_TDM 1
#else
#define USE_TDM 0
#endif

// Load one wave32 16-bit WMMA fragment chunk pair per the CDNA5 ISA layout:
// vector elems 0..7  <-> K = koff..koff+7   (chunk at base)
// vector elems 8..15 <-> K = koff+16..+23   (chunk at base+16)
__device__ __forceinline__ v16bf load_frag(const __bf16* base) {
  v8bf lo = *(const v8bf*)(base);
  v8bf hi = *(const v8bf*)(base + 16);
  return __builtin_shufflevector(lo, hi, 0,1,2,3,4,5,6,7,8,9,10,11,12,13,14,15);
}

#if USE_TDM
// Issue a TDM 2D tile load (bf16): rows x tileW tile from a row-major
// [*, tensorW] matrix into LDS at ldsOff. D# packing per CDNA5 ISA ch.8.
__device__ __forceinline__ void tdm_load_tile_bf16(const __bf16* gsrc, unsigned ldsOff,
                                                   unsigned rows, unsigned tileW,
                                                   unsigned tensorW, unsigned tensorH) {
  unsigned long long ga = (unsigned long long)(size_t)gsrc;
  u32x4 g0;
  g0[0] = 1u;                                      // count=1 (valid), user mode
  g0[1] = ldsOff;                                  // lds_addr (bytes)
  g0[2] = (unsigned)(ga & 0xFFFFFFFFu);            // global_addr[31:0]
  g0[3] = (unsigned)((ga >> 32) & 0x01FFFFFFu)     // global_addr[56:32]
          | 0x80000000u;                           // type=2 ("image") in bits 127:126
  i32x8 g1;
  g1[0] = 0x00010000;                              // wg_mask=0, data_size=1 (2 bytes)
  g1[1] = (int)((tensorW & 0xFFFFu) << 16);        // tensor_dim0[15:0] @ bits 63:48
  g1[2] = (int)(((tensorW >> 16) & 0xFFFFu)        // tensor_dim0[31:16]
          | ((tensorH & 0xFFFFu) << 16));          // tensor_dim1[15:0]
  g1[3] = (int)(((tensorH >> 16) & 0xFFFFu)        // tensor_dim1[31:16]
          | (tileW << 16));                        // tile_dim0 (elements)
  g1[4] = (int)(rows & 0xFFFFu);                   // tile_dim1 = rows, tile_dim2 = 0
  g1[5] = (int)tensorW;                            // tensor_dim0_stride[31:0]
  g1[6] = 0;                                       // stride[47:32]=0, dim1_stride lo=0
  g1[7] = 0;
  i32x4 z4 = {0, 0, 0, 0};
#if __clang_major__ >= 23
  i32x8 z8 = {0, 0, 0, 0, 0, 0, 0, 0};
  __builtin_amdgcn_tensor_load_to_lds(g0, g1, z4, z4, z8, 0);
#else
  __builtin_amdgcn_tensor_load_to_lds(g0, g1, z4, z4, 0);
#endif
}
#endif

// ---------------------------------------------------------------- conversions
__global__ __launch_bounds__(256) void f32_to_bf16_kernel(const float* __restrict__ in,
                                                          __bf16* __restrict__ out, int n) {
  int i = blockIdx.x * 256 + threadIdx.x;
  if (i < n) out[i] = (__bf16)in[i];
}

// V [B*S, D] bf16 -> Vt [B*H, HD, S] bf16 (per-head transpose)
__global__ __launch_bounds__(256) void transpose_v_kernel(const __bf16* __restrict__ V,
                                                          __bf16* __restrict__ Vt) {
  int i = blockIdx.x * 256 + threadIdx.x;           // over B*S*D
  int d = i % Dm;
  int s = (i / Dm) % Ss;
  int b = i / (Dm * Ss);
  int h = d / HDd, dd = d % HDd;
  Vt[(((size_t)(b * Hh + h)) * HDd + dd) * Ss + s] = V[(size_t)i];
}

// -------------------------------------------------------- tiled WMMA GEMM
// Y[M,N] = act(X[M,K] * W[N,K]^T + bias[N]); X,W bf16; out fp32 or bf16.
// Workgroup tile 128(M) x 64(N), KSTEP=64; 8 waves in a 4x2 grid, each wave
// owns a 32x32 block = 2x2 WMMA tiles (8 WMMAs per K-step). K-slices are
// staged in double-buffered LDS by the Tensor Data Mover, software-pipelined
// against the WMMAs using counted s_wait_tensorcnt.
__device__ __forceinline__ void gemm_compute_step(const __bf16* smA, const __bf16* smB,
                                                  int wm, int wn, int mrow, int koff,
                                                  v8f (&c)[2][2]) {
#pragma unroll
  for (int kc = 0; kc < 2; ++kc) {
    v16bf af[2], bf[2];
#pragma unroll
    for (int i = 0; i < 2; ++i)
      af[i] = load_frag(&smA[(wm * 32 + i * 16 + mrow) * 64 + kc * 32 + koff]);
#pragma unroll
    for (int j = 0; j < 2; ++j)
      bf[j] = load_frag(&smB[(wn * 32 + j * 16 + mrow) * 64 + kc * 32 + koff]);
#pragma unroll
    for (int i = 0; i < 2; ++i)
#pragma unroll
      for (int j = 0; j < 2; ++j)
        c[i][j] = __builtin_amdgcn_wmma_f32_16x16x32_bf16(false, af[i], false, bf[j],
                                                          (short)0, c[i][j], false, false);
  }
}

template<int ACT, bool OUTBF16>
__global__ __launch_bounds__(256) void gemm_xwT_kernel(const __bf16* __restrict__ A,
                                                       const __bf16* __restrict__ W,
                                                       const float* __restrict__ bias,
                                                       float* __restrict__ outF,
                                                       __bf16* __restrict__ outB,
                                                       int M, int N, int K) {
  __shared__ __align__(16) __bf16 smA[2][128 * 64];   // 32 KB
  __shared__ __align__(16) __bf16 smB[2][64 * 64];    // 16 KB

  const int lane = threadIdx.x & 31;
  const int wave = threadIdx.x >> 5;
  const int wm = wave >> 1;                  // 0..3
  const int wn = wave & 1;                   // 0..1
  const int row0 = blockIdx.y * 128;
  const int col0 = blockIdx.x * 64;
  const int mrow = lane & 15;
  const int koff = (lane >> 4) * 8;

  v8f c[2][2];
#pragma unroll
  for (int j = 0; j < 2; ++j) {
    const float bv = bias[col0 + wn * 32 + j * 16 + (lane & 15)];
#pragma unroll
    for (int i = 0; i < 2; ++i)
#pragma unroll
      for (int r = 0; r < 8; ++r) c[i][j][r] = bv;
  }

#if USE_TDM
  const unsigned ldsA0 = (unsigned)(size_t)&smA[0][0];  // flat LDS addr truncates to offset
  const unsigned ldsB0 = (unsigned)(size_t)&smB[0][0];
  const unsigned bufAstride = 128 * 64 * 2;             // bytes between A buffers
  const unsigned bufBstride = 64 * 64 * 2;

  if (threadIdx.x < 32) {                    // wave 0 drives the TDM: prologue slice 0
    tdm_load_tile_bf16(A + (size_t)row0 * K, ldsA0, 128, 64, (unsigned)K, (unsigned)M);
    tdm_load_tile_bf16(W + (size_t)col0 * K, ldsB0, 64, 64, (unsigned)K, (unsigned)N);
  }
  int buf = 0;
  for (int k0 = 0; k0 < K; k0 += 64, buf ^= 1) {
    if (threadIdx.x < 32) {
      if (k0 + 64 < K) {                     // prefetch next slice into other buffer
        tdm_load_tile_bf16(A + (size_t)row0 * K + k0 + 64, ldsA0 + (buf ^ 1) * bufAstride,
                           128, 64, (unsigned)K, (unsigned)M);
        tdm_load_tile_bf16(W + (size_t)col0 * K + k0 + 64, ldsB0 + (buf ^ 1) * bufBstride,
                           64, 64, (unsigned)K, (unsigned)N);
        __builtin_amdgcn_s_wait_tensorcnt(2);   // current slice's pair has retired
      } else {
        __builtin_amdgcn_s_wait_tensorcnt(0);   // drain for the final slice
      }
    }
    __syncthreads();                         // current buffer visible to all waves
    gemm_compute_step(&smA[buf][0], &smB[buf][0], wm, wn, mrow, koff, c);
    __syncthreads();                         // buffer consumed; safe to overwrite next iter
  }
#else
  for (int k0 = 0; k0 < K; k0 += 64) {
    __syncthreads();
    for (int e = threadIdx.x * 8; e < 128 * 64; e += 256 * 8) {
      int r = e >> 6, cc = e & 63;
      *(v8bf*)&smA[0][e] = *(const v8bf*)&A[(size_t)(row0 + r) * K + k0 + cc];
    }
    for (int e = threadIdx.x * 8; e < 64 * 64; e += 256 * 8) {
      int r = e >> 6, cc = e & 63;
      *(v8bf*)&smB[0][e] = *(const v8bf*)&W[(size_t)(col0 + r) * K + k0 + cc];
    }
    __syncthreads();
    gemm_compute_step(&smA[0][0], &smB[0][0], wm, wn, mrow, koff, c);
  }
#endif

  const int rsub = (lane >> 4) * 8;
#pragma unroll
  for (int i = 0; i < 2; ++i) {
#pragma unroll
    for (int j = 0; j < 2; ++j) {
      const int ncol = col0 + wn * 32 + j * 16 + (lane & 15);
#pragma unroll
      for (int r = 0; r < 8; ++r) {
        float v = c[i][j][r];
        if (ACT == 1) v = 0.5f * v * (1.f + erff(v * 0.70710678118654752f)); // exact GELU
        size_t o = (size_t)(row0 + wm * 32 + i * 16 + rsub + r) * N + ncol;
        if (OUTBF16) outB[o] = (__bf16)v; else outF[o] = v;
      }
    }
  }
}

// --------------------------------------------------- attention score kernel
// scores[bh, q, k] = (Q_bh[q,:] . K_bh[k,:]) / 8 + structure_bias[h]
__global__ __launch_bounds__(256) void attn_scores_kernel(const __bf16* __restrict__ Q,
                                                          const __bf16* __restrict__ Kc,
                                                          const float* __restrict__ sbias,
                                                          float* __restrict__ attn) {
  const int lane = threadIdx.x & 31;
  const int wave = threadIdx.x >> 5;
  const int tk = blockIdx.x;                 // 0..S/16-1
  const int tq = blockIdx.y * 8 + wave;      // 0..S/32-1
  const int bh = blockIdx.z;                 // 0..B*H-1
  const int b = bh / Hh, h = bh % Hh;
  const int mrow = lane & 15;
  const int koff = (lane >> 4) * 8;

  const __bf16* qrow = Q + ((size_t)(b * Ss + tq * 16 + mrow)) * Dm + h * HDd;
  const __bf16* krow = Kc + ((size_t)(b * Ss + tk * 16 + mrow)) * Dm + h * HDd;

  v8f c = {};
#pragma unroll
  for (int k0 = 0; k0 < HDd; k0 += 32) {
    v16bf a = load_frag(qrow + k0 + koff);
    v16bf bb = load_frag(krow + k0 + koff);
    c = __builtin_amdgcn_wmma_f32_16x16x32_bf16(false, a, false, bb, (short)0, c,
                                                false, false);
  }

  const float sb = sbias[h];
  const int kcol  = tk * 16 + (lane & 15);
  const int qbase = tq * 16 + ((lane >> 4) * 8);
#pragma unroll
  for (int r = 0; r < 8; ++r)
    attn[((size_t)bh * Ss + qbase + r) * Ss + kcol] = c[r] * 0.125f + sb;
}

// -------------------------------------------------------------- row softmax
__global__ __launch_bounds__(256) void softmax_rows_kernel(float* __restrict__ attn) {
  __shared__ float red[256];
  const size_t row = blockIdx.x;             // B*H*S rows
  float* p = attn + row * (size_t)Ss;
  const int t = threadIdx.x;

  float mx = -3.0e38f;
  for (int i = t; i < Ss; i += 256) mx = fmaxf(mx, p[i]);
  red[t] = mx; __syncthreads();
  for (int s = 128; s > 0; s >>= 1) { if (t < s) red[t] = fmaxf(red[t], red[t + s]); __syncthreads(); }
  mx = red[0]; __syncthreads();

  float sum = 0.f;
  for (int i = t; i < Ss; i += 256) { float e = expf(p[i] - mx); p[i] = e; sum += e; }
  red[t] = sum; __syncthreads();
  for (int s = 128; s > 0; s >>= 1) { if (t < s) red[t] += red[t + s]; __syncthreads(); }
  const float inv = 1.f / red[0];
  for (int i = t; i < Ss; i += 256) p[i] *= inv;
}

// --------------------------------------------------- context = attn @ V
// One wave produces a full 16 x 64 (q x HD) output block so the huge softmax
// matrix is streamed from HBM exactly once. A = attn (fp32 -> bf16 on the
// fly), B = Vt[bh, d, s] bf16 (pre-transposed so fragments are contiguous).
__global__ __launch_bounds__(256) void attn_context_kernel(const float* __restrict__ attn,
                                                           const __bf16* __restrict__ Vt,
                                                           float* __restrict__ ctx) {
  const int lane = threadIdx.x & 31;
  const int wave = threadIdx.x >> 5;
  const int tq = blockIdx.y * 8 + wave;      // 0..S/16-1
  const int bh = blockIdx.z;
  const int b = bh / Hh, h = bh % Hh;
  const int mrow = lane & 15;
  const int koff = (lane >> 4) * 8;

  const float*  arow  = attn + ((size_t)bh * Ss + tq * 16 + mrow) * Ss;
  const __bf16* vbase = Vt + (size_t)bh * HDd * Ss + (size_t)mrow * Ss;

  v8f c[4] = {};
  for (int k0 = 0; k0 < Ss; k0 += 32) {
    __builtin_prefetch(arow + k0 + 128, 0, 0);     // global_prefetch_b8
    v16bf a;
#pragma unroll
    for (int j = 0; j < 8; ++j) {
      a[j]     = (__bf16)arow[k0 + koff + j];
      a[8 + j] = (__bf16)arow[k0 + 16 + koff + j];
    }
#pragma unroll
    for (int j = 0; j < 4; ++j) {
      v16bf bb = load_frag(vbase + (size_t)(j * 16) * Ss + k0 + koff);
      c[j] = __builtin_amdgcn_wmma_f32_16x16x32_bf16(false, a, false, bb, (short)0, c[j],
                                                     false, false);
    }
  }

  const int qbase = tq * 16 + ((lane >> 4) * 8);
#pragma unroll
  for (int j = 0; j < 4; ++j) {
    const int dcol = j * 16 + (lane & 15);
#pragma unroll
    for (int r = 0; r < 8; ++r)
      ctx[((size_t)(b * Ss) + qbase + r) * Dm + h * HDd + dcol] = c[j][r];
  }
}

// ------------------------------------------------- residual + layernorm
__global__ __launch_bounds__(256) void add_ln_kernel(const float* __restrict__ a,
                                                     const float* __restrict__ res,
                                                     const float* __restrict__ g,
                                                     const float* __restrict__ bt,
                                                     float* __restrict__ outF,
                                                     __bf16* __restrict__ outB) {
  __shared__ float red[256];
  __shared__ float red2[256];
  const size_t row = blockIdx.x;             // B*S rows of D
  const float* pa = a + row * Dm;
  const float* pr = res + row * Dm;
  const int t = threadIdx.x;

  float s = 0.f, s2 = 0.f, vals[4];
#pragma unroll
  for (int i = 0; i < 4; ++i) {
    int idx = t + i * 256;
    float v = pa[idx] + pr[idx];
    vals[i] = v; s += v; s2 += v * v;
  }
  red[t] = s; red2[t] = s2; __syncthreads();
  for (int st = 128; st > 0; st >>= 1) {
    if (t < st) { red[t] += red[t + st]; red2[t] += red2[t + st]; }
    __syncthreads();
  }
  const float mean = red[0] * (1.f / Dm);
  const float var  = red2[0] * (1.f / Dm) - mean * mean;
  const float inv  = rsqrtf(var + EPSc);
#pragma unroll
  for (int i = 0; i < 4; ++i) {
    int idx = t + i * 256;
    float y = (vals[i] - mean) * inv * g[idx] + bt[idx];
    if (outF) outF[row * Dm + idx] = y;
    if (outB) outB[row * Dm + idx] = (__bf16)y;
  }
}

// =========================================================================
extern "C" void kernel_launch(void* const* d_in, const int* in_sizes, int n_in,
                              void* d_out, int out_size, void* d_ws, size_t ws_size,
                              hipStream_t stream) {
  const float* x     = (const float*)d_in[0];
  const float* Wq    = (const float*)d_in[1];
  const float* bq    = (const float*)d_in[2];
  const float* Wk    = (const float*)d_in[3];
  const float* bk    = (const float*)d_in[4];
  const float* Wv    = (const float*)d_in[5];
  const float* bv    = (const float*)d_in[6];
  const float* sbias = (const float*)d_in[7];
  const float* ln1g  = (const float*)d_in[8];
  const float* ln1b  = (const float*)d_in[9];
  const float* ln2g  = (const float*)d_in[10];
  const float* ln2b  = (const float*)d_in[11];
  const float* W1    = (const float*)d_in[12];
  const float* b1    = (const float*)d_in[13];
  const float* W2    = (const float*)d_in[14];
  const float* b2    = (const float*)d_in[15];

  float* outF = (float*)d_out;                          // [B,S,D]
  float* attn = outF + (size_t)Bb * Ss * Dm;            // [B,H,S,S] (scores -> softmax in place)

  // ---- workspace layout ----
  char* ws = (char*)d_ws;
  const size_t nBSD = (size_t)Bb * Ss * Dm;             // 4,194,304
  size_t off = 0;
  __bf16* xb  = (__bf16*)(ws + off); off += nBSD * 2;
  __bf16* Wqb = (__bf16*)(ws + off); off += (size_t)Dm * Dm * 2;
  __bf16* Wkb = (__bf16*)(ws + off); off += (size_t)Dm * Dm * 2;
  __bf16* Wvb = (__bf16*)(ws + off); off += (size_t)Dm * Dm * 2;
  __bf16* W1b = (__bf16*)(ws + off); off += (size_t)FFf * Dm * 2;
  __bf16* W2b = (__bf16*)(ws + off); off += (size_t)Dm * FFf * 2;
  __bf16* Qb  = (__bf16*)(ws + off); off += nBSD * 2;
  __bf16* Kb  = (__bf16*)(ws + off); off += nBSD * 2;
  __bf16* Vb  = (__bf16*)(ws + off); off += nBSD * 2;
  __bf16* Vt  = (__bf16*)(ws + off); off += nBSD * 2;
  float*  ctx = (float*)(ws + off);  off += nBSD * 4;
  float*  hf  = (float*)(ws + off);  off += nBSD * 4;
  __bf16* hb  = (__bf16*)(ws + off); off += nBSD * 2;
  __bf16* f1b = (__bf16*)(ws + off); off += (size_t)Bb * Ss * FFf * 2;
  float*  f2f = (float*)(ws + off);  off += nBSD * 4;

  const int M = Bb * Ss;                                // 4096

  // 1) fp32 -> bf16 conversions
  f32_to_bf16_kernel<<<(int)((nBSD + 255) / 256), 256, 0, stream>>>(x, xb, (int)nBSD);
  f32_to_bf16_kernel<<<(Dm * Dm + 255) / 256, 256, 0, stream>>>(Wq, Wqb, Dm * Dm);
  f32_to_bf16_kernel<<<(Dm * Dm + 255) / 256, 256, 0, stream>>>(Wk, Wkb, Dm * Dm);
  f32_to_bf16_kernel<<<(Dm * Dm + 255) / 256, 256, 0, stream>>>(Wv, Wvb, Dm * Dm);
  f32_to_bf16_kernel<<<(FFf * Dm + 255) / 256, 256, 0, stream>>>(W1, W1b, FFf * Dm);
  f32_to_bf16_kernel<<<(Dm * FFf + 255) / 256, 256, 0, stream>>>(W2, W2b, Dm * FFf);

  // 2) Q/K/V projections (128x64 tiles, pipelined TDM-staged LDS, bf16 out)
  dim3 gproj(Dm / 64, M / 128);
  gemm_xwT_kernel<0, true><<<gproj, 256, 0, stream>>>(xb, Wqb, bq, nullptr, Qb, M, Dm, Dm);
  gemm_xwT_kernel<0, true><<<gproj, 256, 0, stream>>>(xb, Wkb, bk, nullptr, Kb, M, Dm, Dm);
  gemm_xwT_kernel<0, true><<<gproj, 256, 0, stream>>>(xb, Wvb, bv, nullptr, Vb, M, Dm, Dm);

  // 3) per-head transpose of V for the attn@V GEMM
  transpose_v_kernel<<<(int)((nBSD + 255) / 256), 256, 0, stream>>>(Vb, Vt);

  // 4) scores -> attn region of d_out
  attn_scores_kernel<<<dim3(Ss / 16, Ss / 256, Bb * Hh), 256, 0, stream>>>(Qb, Kb, sbias, attn);

  // 5) softmax rows in place (this is the returned attention output)
  softmax_rows_kernel<<<Bb * Hh * Ss, 256, 0, stream>>>(attn);

  // 6) context = attn @ V (one pass over the attention matrix)
  attn_context_kernel<<<dim3(1, Ss / 128, Bb * Hh), 256, 0, stream>>>(attn, Vt, ctx);

  // 7) h = LN(x + context); also emit bf16 copy for FFN
  add_ln_kernel<<<M, 256, 0, stream>>>(x, ctx, ln1g, ln1b, hf, hb);

  // 8) FFN1: gelu(h @ W1^T + b1) -> bf16
  gemm_xwT_kernel<1, true><<<dim3(FFf / 64, M / 128), 256, 0, stream>>>(hb, W1b, b1, nullptr, f1b, M, FFf, Dm);

  // 9) FFN2: f1 @ W2^T + b2 -> fp32
  gemm_xwT_kernel<0, false><<<dim3(Dm / 64, M / 128), 256, 0, stream>>>(f1b, W2b, b2, f2f, nullptr, M, Dm, FFf);

  // 10) out = LN(h + ffn)
  add_ln_kernel<<<M, 256, 0, stream>>>(hf, f2f, ln2g, ln2b, outF, (__bf16*)nullptr);
}
